// SparseLightningIndexerKLLoss_86354612453568
// MI455X (gfx1250) — compile-verified, matrix operand order checked by reading, not run on hardware
//
#include <hip/hip_runtime.h>
#include <hip/hip_bf16.h>
#include <math.h>

typedef __attribute__((ext_vector_type(16))) _Float16 v16h;
typedef __attribute__((ext_vector_type(8)))  _Float16 v8h;
typedef __attribute__((ext_vector_type(8)))  float    v8f;

#define S_LEN   2048
#define K_SEL   512
#define SCALE_F 0.08838834764831845f   // 1/sqrt(128)
#define EPS_F   1e-9f

// ws layout: [0, 8KB) per-t KL partials; [8KB, +1MB) keyCvt; then kidxCvt (+1MB)
#define WS_KL_BYTES   8192
#define CVT_BYTES     (S_LEN * 512)          // per array: 2048 keys * 512B
#define WS_NEED       (WS_KL_BYTES + 2 * CVT_BYTES)

__device__ __forceinline__ v8f wmma_f16(v16h a, v16h b, v8f c) {
  // D = A(16x32 f16) * B(32x16 f16) + C(16x16 f32)
  return __builtin_amdgcn_wmma_f32_16x16x32_f16(false, a, false, b, (short)0, c,
                                                false, false);
}

// Build hi/lo f16 fragment halves for one 32-wide K chunk from f32 row data.
// Element e<8 -> feature [o1+e]; e>=8 -> feature [o2+e-8].
__device__ __forceinline__ void cvt16(const float* __restrict__ p, int o1, int o2,
                                      v16h& hi, v16h& lo) {
  const float4 x0 = *(const float4*)(p + o1);
  const float4 x1 = *(const float4*)(p + o1 + 4);
  const float4 x2 = *(const float4*)(p + o2);
  const float4 x3 = *(const float4*)(p + o2 + 4);
  float f[16] = {x0.x, x0.y, x0.z, x0.w, x1.x, x1.y, x1.z, x1.w,
                 x2.x, x2.y, x2.z, x2.w, x3.x, x3.y, x3.z, x3.w};
#pragma unroll
  for (int e = 0; e < 16; ++e) {
    const _Float16 h = (_Float16)f[e];
    hi[e] = h;
    lo[e] = (_Float16)(f[e] - (float)h);
  }
}

// -------- Pre-pass: convert key / key_indexer rows to fragment-ordered hi/lo f16 ----
// Converted layout (in halves): block b = (s*4 + c)*2 + hf holds
//   [b*32 + 0..15]  = hi elements 0..15 of the (c,hf) fragment slice of key s
//   [b*32 + 16..31] = lo elements 0..15
// Fragment element e maps to feature: e<8 -> 32c+8hf+e ; e>=8 -> 32c+16+8hf+(e-8)
__global__ __launch_bounds__(256)
void sli_cvt_kernel(const float* __restrict__ key,   // [S,128]
                    const float* __restrict__ kidx,  // [S,128]
                    _Float16* __restrict__ keyCvt,   // [S*256] halves
                    _Float16* __restrict__ kidxCvt) {
  const unsigned u = blockIdx.x * 256 + threadIdx.x;   // 65536 total
  const int a  = u >> 15;          // 0 = key, 1 = kidx
  const int s  = (u >> 4) & 2047;  // key row
  const int g  = u & 15;           // group of 8 features
  const int c  = g >> 2;
  const int gg = g & 3;
  const int hf = gg & 1;
  const int eb = (gg >> 1) * 8;    // element base within fragment (0 or 8)
  const int d0 = c * 32 + gg * 8;  // feature base

  const float* src = (a == 0 ? key : kidx) + (size_t)s * 128 + d0;
  _Float16* dst = (a == 0 ? keyCvt : kidxCvt);

  const float4 x0 = *(const float4*)(src);
  const float4 x1 = *(const float4*)(src + 4);
  float f[8] = {x0.x, x0.y, x0.z, x0.w, x1.x, x1.y, x1.z, x1.w};
  v8h hi, lo;
#pragma unroll
  for (int e = 0; e < 8; ++e) {
    const _Float16 h = (_Float16)f[e];
    hi[e] = h;
    lo[e] = (_Float16)(f[e] - (float)h);
  }
  const size_t blk = ((size_t)s * 4 + c) * 2 + hf;   // 32-half block
  *(v8h*)(dst + blk * 32 + eb)      = hi;            // 16B aligned store
  *(v8h*)(dst + blk * 32 + 16 + eb) = lo;
}

// ---------------- Shared epilogue: block reductions + KL ----------------
__device__ __forceinline__ void kl_epilogue(float* s_p, float* s_I,
                                            float* s_redA, float* s_redB,
                                            float* __restrict__ ws, int t) {
  const int tid  = threadIdx.x;
  const int wave = tid >> 5;
  const int lane = tid & 31;

  // Pass A: Psum, Imax
  float lp = s_p[tid] + s_p[tid + 256];
  float li = fmaxf(s_I[tid], s_I[tid + 256]);
#pragma unroll
  for (int o = 16; o >= 1; o >>= 1) {
    lp += __shfl_xor(lp, o, 32);
    li = fmaxf(li, __shfl_xor(li, o, 32));
  }
  if (lane == 0) { s_redA[wave] = lp; s_redB[wave] = li; }
  __syncthreads();
  float Psum = 0.0f, Imax = -3.0e38f;
#pragma unroll
  for (int w = 0; w < 8; ++w) {
    Psum += s_redA[w];
    Imax = fmaxf(Imax, s_redB[w]);
  }
  __syncthreads();

  // Pass B: Esum
  float le = expf(s_I[tid] - Imax) + expf(s_I[tid + 256] - Imax);
#pragma unroll
  for (int o = 16; o >= 1; o >>= 1) le += __shfl_xor(le, o, 32);
  if (lane == 0) s_redA[wave] = le;
  __syncthreads();
  float Esum = 0.0f;
#pragma unroll
  for (int w = 0; w < 8; ++w) Esum += s_redA[w];
  const float lZ   = logf(Esum);
  const float invP = 1.0f / (Psum + EPS_F);

  // Pass C: KL terms
  float lk = 0.0f;
#pragma unroll
  for (int jj = 0; jj < 2; ++jj) {
    const int j = tid + jj * 256;
    const float pn = s_p[j] * invP;
    const float lq = s_I[j] - Imax - lZ;
    lk += pn * (logf(pn + EPS_F) - lq);
  }
#pragma unroll
  for (int o = 16; o >= 1; o >>= 1) lk += __shfl_xor(lk, o, 32);
  if (lane == 0) s_redB[wave] = lk;
  __syncthreads();
  if (tid == 0) {
    float tt = 0.0f;
#pragma unroll
    for (int w = 0; w < 8; ++w) tt += s_redB[w];
    ws[t] = tt;
  }
}

// ---------------- Fast main kernel: LDS-staged A, pre-converted B ----------------
__global__ __launch_bounds__(256)
void sli_kl_fast(const float* __restrict__ query,      // [S,16,128]
                 const float* __restrict__ qidx,       // [S,8,128]
                 const _Float16* __restrict__ keyCvt,  // fragment-ordered hi/lo
                 const _Float16* __restrict__ kidxCvt,
                 const float* __restrict__ weights,    // [S,8]
                 const int*   __restrict__ sidx,       // [S,512]
                 const float* __restrict__ smax,       // [16,S]
                 const float* __restrict__ ssum,       // [16,S]
                 float* __restrict__ ws) {
  const int t    = blockIdx.x;
  const int tid  = threadIdx.x;
  const int wave = tid >> 5;
  const int lane = tid & 31;
  const int n    = lane & 15;
  const int hf   = lane >> 4;

  __shared__ v16h  s_afrag[16][32];   // fragid = (mat*4 + c)*2 + part; 16KB
  __shared__ float s_mx[16], s_rinv[16], s_w[8];
  __shared__ int   s_idx[K_SEL];
  __shared__ float s_p[K_SEL], s_I[K_SEL];
  __shared__ float s_redA[8], s_redB[8];

  if (tid < 16) {
    s_mx[tid]   = smax[tid * S_LEN + t];
    s_rinv[tid] = 1.0f / ssum[tid * S_LEN + t];
  }
  if (tid < 8) s_w[tid] = weights[t * 8 + tid];
  s_idx[tid]       = sidx[(size_t)t * K_SEL + tid];
  s_idx[tid + 256] = sidx[(size_t)t * K_SEL + tid + 256];

  // Cooperatively stage the 16 A fragments: wave w -> (mat = w>>2, c = w&3)
  {
    const int mat = wave >> 2;
    const int c   = wave & 3;
    const int o1  = c * 32 + 8 * hf;
    const int o2  = o1 + 16;
    v16h hi = {}, lo = {};
    if (mat == 0) {
      cvt16(query + ((size_t)t * 16 + n) * 128, o1, o2, hi, lo);
    } else if (n < 8) {
      cvt16(qidx + ((size_t)t * 8 + n) * 128, o1, o2, hi, lo);
    }
    const int fb = (mat * 4 + c) * 2;
    s_afrag[fb + 0][lane] = hi;
    s_afrag[fb + 1][lane] = lo;
  }
  __syncthreads();

  const v16h* kB = (const v16h*)keyCvt;    // [ (s*4+c)*2+hf ][2] v16h (hi,lo)
  const v16h* iB = (const v16h*)kidxCvt;

  for (int tl = 0; tl < 4; ++tl) {
    const int kb = (wave * 4 + tl) * 16;
    const int s  = s_idx[kb + n];
    const size_t bbase = (((size_t)s * 4) * 2 + hf) * 2;  // v16h index at c=0

    v8f acc0 = {};
    v8f acc1 = {};
#pragma unroll
    for (int c = 0; c < 4; ++c) {
      const v16h Ah = s_afrag[c * 2 + 0][lane];
      const v16h Al = s_afrag[c * 2 + 1][lane];
      const v16h Ih = s_afrag[8 + c * 2 + 0][lane];
      const v16h Il = s_afrag[8 + c * 2 + 1][lane];
      const size_t bo = bbase + (size_t)c * 4;   // +c*(2 hf * 2 v16h)
      const v16h Bh = kB[bo + 0];
      const v16h Bl = kB[bo + 1];
      const v16h Ch = iB[bo + 0];
      const v16h Cl = iB[bo + 1];
      acc0 = wmma_f16(Ah, Bh, acc0);
      acc0 = wmma_f16(Ah, Bl, acc0);
      acc0 = wmma_f16(Al, Bh, acc0);
      acc1 = wmma_f16(Ih, Ch, acc1);
      acc1 = wmma_f16(Ih, Cl, acc1);
      acc1 = wmma_f16(Il, Ch, acc1);
    }

    float ps = 0.0f;
#pragma unroll
    for (int r = 0; r < 8; ++r) {
      const int mm = r + 8 * hf;
      ps += expf(SCALE_F * acc0[r] - s_mx[mm]) * s_rinv[mm];
    }
    ps += __shfl_xor(ps, 16, 32);

    float Ic = 0.0f;
#pragma unroll
    for (int r = 0; r < 8; ++r) Ic += s_w[r] * fmaxf(acc1[r], 0.0f);

    if (hf == 0) {
      s_p[kb + n] = ps;
      s_I[kb + n] = Ic;
    }
  }
  __syncthreads();
  kl_epilogue(s_p, s_I, s_redA, s_redB, ws, t);
}

// ---------------- Fallback main kernel: convert-in-loop (small ws) ----------------
__global__ __launch_bounds__(256)
void sli_kl_fallback(const float* __restrict__ query,
                     const float* __restrict__ key,
                     const float* __restrict__ qidx,
                     const float* __restrict__ kidx,
                     const float* __restrict__ weights,
                     const int*   __restrict__ sidx,
                     const float* __restrict__ smax,
                     const float* __restrict__ ssum,
                     float* __restrict__ ws) {
  const int t    = blockIdx.x;
  const int tid  = threadIdx.x;
  const int wave = tid >> 5;
  const int lane = tid & 31;
  const int n    = lane & 15;
  const int hf   = lane >> 4;

  __shared__ float s_mx[16], s_rinv[16], s_w[8];
  __shared__ int   s_idx[K_SEL];
  __shared__ float s_p[K_SEL], s_I[K_SEL];
  __shared__ float s_redA[8], s_redB[8];

  if (tid < 16) {
    s_mx[tid]   = smax[tid * S_LEN + t];
    s_rinv[tid] = 1.0f / ssum[tid * S_LEN + t];
  }
  if (tid < 8) s_w[tid] = weights[t * 8 + tid];
  s_idx[tid]       = sidx[(size_t)t * K_SEL + tid];
  s_idx[tid + 256] = sidx[(size_t)t * K_SEL + tid + 256];
  __syncthreads();

  const float* qArow = query + ((size_t)t * 16 + n) * 128;
  const float* qIrow = qidx  + ((size_t)t * 8 + (n & 7)) * 128;

  for (int tl = 0; tl < 4; ++tl) {
    const int kb = (wave * 4 + tl) * 16;
    const int s  = s_idx[kb + n];
    const float* kArow = key  + (size_t)s * 128;
    const float* kIrow = kidx + (size_t)s * 128;

    v8f acc0 = {};
    v8f acc1 = {};
#pragma unroll
    for (int c = 0; c < 4; ++c) {
      const int o1 = c * 32 + 8 * hf;
      const int o2 = o1 + 16;
      v16h Ah, Al, Ih, Il, Bh, Bl, Ch, Cl;
      cvt16(qArow, o1, o2, Ah, Al);
      if (n < 8) {
        cvt16(qIrow, o1, o2, Ih, Il);
      } else {
        Ih = (v16h){};
        Il = (v16h){};
      }
      cvt16(kArow, o1, o2, Bh, Bl);
      cvt16(kIrow, o1, o2, Ch, Cl);
      acc0 = wmma_f16(Ah, Bh, acc0);
      acc0 = wmma_f16(Ah, Bl, acc0);
      acc0 = wmma_f16(Al, Bh, acc0);
      acc1 = wmma_f16(Ih, Ch, acc1);
      acc1 = wmma_f16(Ih, Cl, acc1);
      acc1 = wmma_f16(Il, Ch, acc1);
    }

    float ps = 0.0f;
#pragma unroll
    for (int r = 0; r < 8; ++r) {
      const int mm = r + 8 * hf;
      ps += expf(SCALE_F * acc0[r] - s_mx[mm]) * s_rinv[mm];
    }
    ps += __shfl_xor(ps, 16, 32);

    float Ic = 0.0f;
#pragma unroll
    for (int r = 0; r < 8; ++r) Ic += s_w[r] * fmaxf(acc1[r], 0.0f);

    if (hf == 0) {
      s_p[kb + n] = ps;
      s_I[kb + n] = Ic;
    }
  }
  __syncthreads();
  kl_epilogue(s_p, s_I, s_redA, s_redB, ws, t);
}

// Deterministic fixed-tree final reduction: out = sum_t ws[t] / S
__global__ __launch_bounds__(256)
void sli_reduce_kernel(const float* __restrict__ ws, float* __restrict__ out) {
  __shared__ float red[8];
  const int tid = threadIdx.x;
  float s = 0.0f;
  for (int j = tid; j < S_LEN; j += 256) s += ws[j];
#pragma unroll
  for (int o = 16; o >= 1; o >>= 1) s += __shfl_xor(s, o, 32);
  if ((tid & 31) == 0) red[tid >> 5] = s;
  __syncthreads();
  if (tid == 0) {
    float tt = 0.0f;
#pragma unroll
    for (int w = 0; w < 8; ++w) tt += red[w];
    out[0] = tt * (1.0f / (float)S_LEN);
  }
}

extern "C" void kernel_launch(void* const* d_in, const int* in_sizes, int n_in,
                              void* d_out, int out_size, void* d_ws, size_t ws_size,
                              hipStream_t stream) {
  const float* query   = (const float*)d_in[0];
  const float* key     = (const float*)d_in[1];
  const float* qidx    = (const float*)d_in[2];
  const float* kidx    = (const float*)d_in[3];
  const float* weights = (const float*)d_in[4];
  const int*   sidx    = (const int*)d_in[5];
  const float* smax    = (const float*)d_in[6];
  const float* ssum    = (const float*)d_in[7];
  float* out = (float*)d_out;

  float*    ws_kl   = (float*)d_ws;
  _Float16* keyCvt  = (_Float16*)((char*)d_ws + WS_KL_BYTES);
  _Float16* kidxCvt = (_Float16*)((char*)d_ws + WS_KL_BYTES + CVT_BYTES);

  if (ws_size >= (size_t)WS_NEED) {
    sli_cvt_kernel<<<256, 256, 0, stream>>>(key, kidx, keyCvt, kidxCvt);
    sli_kl_fast<<<S_LEN, 256, 0, stream>>>(query, qidx, keyCvt, kidxCvt,
                                           weights, sidx, smax, ssum, ws_kl);
  } else {
    sli_kl_fallback<<<S_LEN, 256, 0, stream>>>(query, key, qidx, kidx, weights,
                                               sidx, smax, ssum, ws_kl);
  }
  sli_reduce_kernel<<<1, 256, 0, stream>>>(ws_kl, out);
}
